// CustomMetalPKA_GNN_88914412961903
// MI455X (gfx1250) — compile-verified, matrix operand order checked by reading, not run on hardware
//
#include <hip/hip_runtime.h>
#include <hip/hip_bf16.h>

#define N_NODES   9999
#define MPAD      10000          // 625 tiles of 16 rows
#define E_EDGES   160000
#define NODE_D    256
#define HID       512
#define HEADS     8
#define HD        64
#define MAX_LIG   3
#define APL       3333           // N // MAX_LIG
#define METAL_EMB 64
#define NBLK      4              // 16x64 output per wave
#define ECHUNK    128            // edges staged in LDS per gather chunk

typedef __attribute__((ext_vector_type(16))) __bf16          v16bf;
typedef __attribute__((ext_vector_type(8)))  float           v8f;
typedef __attribute__((ext_vector_type(8)))  unsigned short  v8u;
typedef __attribute__((ext_vector_type(16))) unsigned short  v16u;

__device__ __forceinline__ unsigned short f2bf(float f) {
  unsigned u = __float_as_uint(f);
  u += 0x7FFFu + ((u >> 16) & 1u);          // round-to-nearest-even
  return (unsigned short)(u >> 16);
}

// ---------------- CSR build: histogram -> scan -> bucket ----------------
__global__ void k_cnt_init(int* cnt) {
  int i = blockIdx.x * blockDim.x + threadIdx.x;
  if (i < MPAD) cnt[i] = 0;
}
__global__ void k_cnt_edges(const int* __restrict__ dst, int* cnt) {
  int e = blockIdx.x * blockDim.x + threadIdx.x;
  if (e < E_EDGES) atomicAdd(&cnt[dst[e]], 1);
}
__global__ void k_dinv(const int* __restrict__ cnt, float* dinv) {
  int i = blockIdx.x * blockDim.x + threadIdx.x;
  if (i < MPAD) dinv[i] = (i < N_NODES) ? rsqrtf((float)(cnt[i] + 1)) : 0.0f; // +1 self-loop
}
#define SCAN_T 512
__global__ void k_scan(const int* __restrict__ cnt, int* __restrict__ off, int* __restrict__ cur) {
  __shared__ int part[SCAN_T];
  int t = threadIdx.x;
  const int CH = (MPAD + SCAN_T - 1) / SCAN_T;             // 20
  int base = t * CH;
  int s = 0;
  for (int i = 0; i < CH; ++i) { int idx = base + i; if (idx < MPAD) s += cnt[idx]; }
  part[t] = s;
  __syncthreads();
  for (int d = 1; d < SCAN_T; d <<= 1) {                   // Hillis-Steele inclusive scan
    int v = (t >= d) ? part[t - d] : 0;
    __syncthreads();
    part[t] += v;
    __syncthreads();
  }
  int run = (t == 0) ? 0 : part[t - 1];                    // exclusive base of this chunk
  for (int i = 0; i < CH; ++i) {
    int idx = base + i;
    if (idx < MPAD) { off[idx] = run; cur[idx] = run; run += cnt[idx]; }
  }
  if (t == SCAN_T - 1) off[MPAD] = run;                    // == E
}
__global__ void k_bucket(const int* __restrict__ src, const int* __restrict__ dst,
                         const float* __restrict__ dinv,
                         int* cur, int* __restrict__ srcid, float* __restrict__ wnorm) {
  int e = blockIdx.x * blockDim.x + threadIdx.x;
  if (e >= E_EDGES) return;
  int s = src[e], d = dst[e];
  int pos = atomicAdd(&cur[d], 1);
  srcid[pos] = s;
  wnorm[pos] = dinv[s] * dinv[d];
}

// ---------------- bf16 prep ----------------
__global__ void k_cast_x(const float* __restrict__ x, unsigned short* __restrict__ xb) {
  int i = blockIdx.x * blockDim.x + threadIdx.x;
  if (i >= MPAD * NODE_D) return;
  int r = i / NODE_D;
  xb[i] = (r < N_NODES) ? f2bf(x[i]) : (unsigned short)0;
}
__global__ void k_transpose_w(const float* __restrict__ w, unsigned short* __restrict__ wt,
                              int K, int Nout) {
  int i = blockIdx.x * blockDim.x + threadIdx.x;
  if (i >= K * Nout) return;
  int k = i / Nout, n = i % Nout;
  wt[n * K + k] = f2bf(w[i]);               // N-major for WMMA B fragments
}

// -------- WMMA GEMM: C[M,512] = A[M,K](bf16) * BT[512,K](bf16); 16x64 per wave --------
template<int K>
__global__ void k_gemm_wmma(const unsigned short* __restrict__ A,
                            const unsigned short* __restrict__ BT,
                            float* __restrict__ C,
                            const float* __restrict__ bias,
                            int NB, int totalWTiles, int relu)
{
  int wid = blockIdx.x * (blockDim.x >> 5) + (threadIdx.x >> 5);
  if (wid >= totalWTiles) return;
  int lane = threadIdx.x & 31;
  int half = lane >> 4;                      // K-half selector (16-bit WMMA layout)
  int lm   = lane & 15;                      // row (A) / col (B) within tile
  int mt = wid / NB, nb = wid % NB;
  int mbase = mt * 16, nbase = nb * (16 * NBLK);

  const unsigned short* arow = A + (size_t)(mbase + lm) * K + half * 8;
  const unsigned short* brow[NBLK];
#pragma unroll
  for (int s = 0; s < NBLK; ++s)
    brow[s] = BT + (size_t)(nbase + s * 16 + lm) * K + half * 8;

  v8f acc[NBLK];
#pragma unroll
  for (int s = 0; s < NBLK; ++s) acc[s] = (v8f){0.f,0.f,0.f,0.f,0.f,0.f,0.f,0.f};

#pragma unroll
  for (int k0 = 0; k0 < K; k0 += 32) {
    v8u a0 = *(const v8u*)(arow + k0);
    v8u a1 = *(const v8u*)(arow + k0 + 16);
    v16bf av = __builtin_bit_cast(v16bf,
        __builtin_shufflevector(a0, a1, 0,1,2,3,4,5,6,7,8,9,10,11,12,13,14,15));
    v16bf bv[NBLK];
#pragma unroll
    for (int s = 0; s < NBLK; ++s) {         // issue all B loads of this k-step first
      v8u b0 = *(const v8u*)(brow[s] + k0);
      v8u b1 = *(const v8u*)(brow[s] + k0 + 16);
      bv[s] = __builtin_bit_cast(v16bf,
          __builtin_shufflevector(b0, b1, 0,1,2,3,4,5,6,7,8,9,10,11,12,13,14,15));
    }
#pragma unroll
    for (int s = 0; s < NBLK; ++s)
      acc[s] = __builtin_amdgcn_wmma_f32_16x16x32_bf16(false, av, false, bv[s],
                                                       (short)0, acc[s], false, false);
  }

#pragma unroll
  for (int s = 0; s < NBLK; ++s) {
    int col = nbase + s * 16 + lm;
    float bi = bias ? bias[col] : 0.0f;
    float* out = C + (size_t)(mbase + half * 8) * HID + col;  // D layout: VGPR r -> M = half*8 + r
#pragma unroll
    for (int r = 0; r < 8; ++r) {
      float v = acc[s][r] + bi;
      if (relu) v = fmaxf(v, 0.0f);
      out[(size_t)r * HID] = v;
    }
  }
}

// ------- GCN aggregation: per-dst gather + self-loop + bias + ReLU + bf16 cast -------
__global__ void k_gather(const float* __restrict__ HL, const int* __restrict__ off,
                         const int* __restrict__ srcid, const float* __restrict__ wnorm,
                         const float* __restrict__ dinv, const float* __restrict__ bias,
                         unsigned short* __restrict__ HB) {
  __shared__ int   s_src[ECHUNK];
  __shared__ float s_w[ECHUNK];
  int n = blockIdx.x;                        // MPAD blocks
  int c = threadIdx.x;                       // 512 threads = HID channels
  if (n >= N_NODES) { HB[(size_t)n * HID + c] = 0; return; }
  float di = dinv[n];
  float acc = HL[(size_t)n * HID + c] * di * di + bias[c];   // self-loop + bias
  int j0 = off[n], j1 = off[n + 1];
  for (int base = j0; base < j1; base += ECHUNK) {
    int m = j1 - base; if (m > ECHUNK) m = ECHUNK;
    if (c < m) { s_src[c] = srcid[base + c]; s_w[c] = wnorm[base + c]; }
    __syncthreads();
    for (int t = 0; t < m; ++t)
      acc += HL[(size_t)s_src[t] * HID + c] * s_w[t];
    __syncthreads();
  }
  HB[(size_t)n * HID + c] = f2bf(fmaxf(acc, 0.0f));
}

// ---------------- backbone means ----------------
__global__ void k_backbone(const float* __restrict__ H1, const int* __restrict__ pred_pos,
                           float* __restrict__ bb) {
  int i = blockIdx.x;                        // ligand 0..2
  int c = threadIdx.x;                       // 512
  int idx = pred_pos[i];
  int start = (idx / APL) * APL;
  float s = 0.0f;
  for (int r = 0; r < APL; ++r) s += H1[(size_t)(start + r) * HID + c];
  bb[i * HID + c] = (s - H1[(size_t)idx * HID + c]) / (float)(APL - 1);
}

// ---------------- metal node + gates + TransformerConv + heads ----------------
__global__ void k_tail(const float* __restrict__ H1, const float* __restrict__ bb,
                       const int* __restrict__ metal_id, const int* __restrict__ pred_pos,
                       const float* __restrict__ emb_table,
                       const float* __restrict__ mp_w, const float* __restrict__ mp_b,
                       const float* __restrict__ gw1, const float* __restrict__ gb1,
                       const float* __restrict__ gw2, const float* __restrict__ gb2,
                       const float* __restrict__ tq_w, const float* __restrict__ tq_b,
                       const float* __restrict__ tk_w, const float* __restrict__ tk_b,
                       const float* __restrict__ tv_w, const float* __restrict__ tv_b,
                       const float* __restrict__ tsk_w, const float* __restrict__ tsk_b,
                       const float* __restrict__ pr_w1, const float* __restrict__ pr_b1,
                       const float* __restrict__ pr_w2, const float* __restrict__ pr_b2,
                       float* __restrict__ out)
{
  __shared__ float s_hm[4 * HID];
  __shared__ float s_q[4 * HID], s_k[4 * HID], s_v[4 * HID], s_sk[4 * HID];
  __shared__ float s_V[HID];
  __shared__ float s_t[256];
  __shared__ float s_alpha[MAX_LIG * HEADS];
  __shared__ float s_att[MAX_LIG * HEADS];
  __shared__ float s_scalar;
  int c = threadIdx.x;                       // 512 threads

  // metal node = relu(emb @ mp_w + mp_b) -> hm[0]
  int mid = metal_id[0];
  {
    float s = mp_b[c];
    for (int j = 0; j < METAL_EMB; ++j) s += emb_table[mid * METAL_EMB + j] * mp_w[j * HID + c];
    s_hm[c] = fmaxf(s, 0.0f);
  }
  __syncthreads();

  // gated binding-site rows -> hm[1..3]
  for (int i = 0; i < MAX_LIG; ++i) {
    if (c < 256) {
      float s = gb1[c];
      for (int j = 0; j < HID; ++j) s += bb[i * HID + j] * gw1[j * 256 + c];
      s_t[c] = tanhf(s);
    }
    __syncthreads();
    if (c == 0) {
      float s = gb2[0];
      for (int j = 0; j < 256; ++j) s += s_t[j] * gw2[j];
      s_scalar = 1.0f / (1.0f + expf(-s));
    }
    __syncthreads();
    float g = s_scalar;
    int idx = pred_pos[i];
    s_hm[(i + 1) * HID + c] = H1[(size_t)idx * HID + c] + g * bb[i * HID + c];
    __syncthreads();
  }

  // q,k,v,skip for all 4 rows (shared across the 3 prediction passes)
  for (int r = 0; r < 4; ++r) {
    float aq = tq_b[c], ak = tk_b[c], av = tv_b[c], as = tsk_b[c];
    for (int j = 0; j < HID; ++j) {
      float hj = s_hm[r * HID + j];
      aq += hj * tq_w[j * HID + c];
      ak += hj * tk_w[j * HID + c];
      av += hj * tv_w[j * HID + c];
      as += hj * tsk_w[j * HID + c];
    }
    s_q[r * HID + c] = aq; s_k[r * HID + c] = ak;
    s_v[r * HID + c] = av; s_sk[r * HID + c] = as;
  }
  __syncthreads();

  // alpha for edges (l -> 0): q[0].k[l] per head (eattr == 0)
  if (c < MAX_LIG * HEADS) {
    int l = c / HEADS + 1, h = c % HEADS;
    float s = 0.0f;
    for (int d = 0; d < HD; ++d) s += s_q[h * HD + d] * s_k[l * HID + h * HD + d];
    s_alpha[c] = s * 0.125f;                 // 1/sqrt(64)
  }
  __syncthreads();

  for (int nl = MAX_LIG; nl >= 1; --nl) {
    if (c < HEADS) {
      float amax = -3.4e38f;
      for (int l = 0; l < nl; ++l) amax = fmaxf(amax, s_alpha[l * HEADS + c]);
      float den = 0.0f;
      for (int l = 0; l < nl; ++l) {
        float e = expf(s_alpha[l * HEADS + c] - amax);
        s_att[l * HEADS + c] = e; den += e;
      }
      for (int l = 0; l < nl; ++l) s_att[l * HEADS + c] /= den;
    }
    __syncthreads();
    {
      int h = c / HD;
      float o = 0.0f;
      for (int l = 0; l < nl; ++l) o += s_att[l * HEADS + h] * s_v[(l + 1) * HID + c];
      float vsum = o + s_sk[c];              // h3[0]
      for (int l = 1; l <= nl; ++l) vsum += s_v[c] + s_sk[l * HID + c];  // h3[l] = v[0]+skip[l]
      s_V[c] = vsum / (float)(nl + 1);
    }
    __syncthreads();
    if (c < 256) {
      float s = pr_b1[c];
      for (int j = 0; j < HID; ++j) s += s_V[j] * pr_w1[j * 256 + c];
      s_t[c] = s;
    }
    __syncthreads();
    if (c == 0) {
      float s = pr_b2[0];
      for (int j = 0; j < 256; ++j) s += s_t[j] * pr_w2[j];
      out[MAX_LIG - nl] = s;
    }
    __syncthreads();
  }
}

// ---------------- launch ----------------
extern "C" void kernel_launch(void* const* d_in, const int* in_sizes, int n_in,
                              void* d_out, int out_size, void* d_ws, size_t ws_size,
                              hipStream_t stream)
{
  (void)in_sizes; (void)n_in; (void)out_size; (void)ws_size;
  const float* x        = (const float*)d_in[0];
  const int*   ei       = (const int*)  d_in[1];
  const int*   src      = ei;
  const int*   dst      = ei + E_EDGES;
  const int*   metal_id = (const int*)  d_in[2];
  const int*   pred_pos = (const int*)  d_in[3];
  const float* emb      = (const float*)d_in[4];
  const float* mp_w     = (const float*)d_in[5];
  const float* mp_b     = (const float*)d_in[6];
  const float* g1_w     = (const float*)d_in[7];
  const float* g1_b     = (const float*)d_in[8];
  const float* g2_w     = (const float*)d_in[9];
  const float* g2_b     = (const float*)d_in[10];
  const float* lp_w     = (const float*)d_in[11];
  const float* lp_b     = (const float*)d_in[12];
  const float* gw1      = (const float*)d_in[13];
  const float* gb1      = (const float*)d_in[14];
  const float* gw2      = (const float*)d_in[15];
  const float* gb2      = (const float*)d_in[16];
  const float* tq_w     = (const float*)d_in[17];
  const float* tq_b     = (const float*)d_in[18];
  const float* tk_w     = (const float*)d_in[19];
  const float* tk_b     = (const float*)d_in[20];
  const float* tv_w     = (const float*)d_in[21];
  const float* tv_b     = (const float*)d_in[22];
  /* te_w = d_in[23] unused: eattr == 0 */
  const float* tsk_w    = (const float*)d_in[24];
  const float* tsk_b    = (const float*)d_in[25];
  const float* pr_w1    = (const float*)d_in[26];
  const float* pr_b1    = (const float*)d_in[27];
  const float* pr_w2    = (const float*)d_in[28];
  const float* pr_b2    = (const float*)d_in[29];

  char* ws = (char*)d_ws;
  auto carve = [&](size_t bytes) -> char* {
    char* p = ws; ws += (bytes + 255) & ~(size_t)255; return p;
  };
  unsigned short* Xbf   = (unsigned short*)carve((size_t)MPAD * NODE_D * 2);
  unsigned short* W1T   = (unsigned short*)carve((size_t)HID  * NODE_D * 2);
  unsigned short* W2T   = (unsigned short*)carve((size_t)HID  * HID    * 2);
  unsigned short* WLT   = (unsigned short*)carve((size_t)HID  * HID    * 2);
  float*          HL    = (float*)         carve((size_t)MPAD * HID * 4);
  float*          HA    = (float*)         carve((size_t)MPAD * HID * 4);   // H1 (fp32)
  unsigned short* HB    = (unsigned short*)carve((size_t)MPAD * HID * 2);
  int*            cnt   = (int*)           carve((size_t)MPAD * 4);
  int*            off   = (int*)           carve((size_t)(MPAD + 1) * 4);
  int*            cur   = (int*)           carve((size_t)MPAD * 4);
  float*          dnv   = (float*)         carve((size_t)MPAD * 4);
  int*            srcid = (int*)           carve((size_t)E_EDGES * 4);
  float*          wnorm = (float*)         carve((size_t)E_EDGES * 4);
  float*          bbf   = (float*)         carve((size_t)MAX_LIG * HID * 4);

  // CSR build (shared by both GCN layers) + dinv
  k_cnt_init <<<(MPAD + 255) / 256, 256, 0, stream>>>(cnt);
  k_cnt_edges<<<(E_EDGES + 255) / 256, 256, 0, stream>>>(dst, cnt);
  k_scan     <<<1, SCAN_T, 0, stream>>>(cnt, off, cur);
  k_dinv     <<<(MPAD + 255) / 256, 256, 0, stream>>>(cnt, dnv);
  k_bucket   <<<(E_EDGES + 255) / 256, 256, 0, stream>>>(src, dst, dnv, cur, srcid, wnorm);

  // bf16 prep
  k_cast_x     <<<(MPAD * NODE_D + 255) / 256, 256, 0, stream>>>(x, Xbf);
  k_transpose_w<<<(NODE_D * HID + 255) / 256, 256, 0, stream>>>(g1_w, W1T, NODE_D, HID);
  k_transpose_w<<<(HID * HID + 255) / 256, 256, 0, stream>>>(g2_w, W2T, HID, HID);
  k_transpose_w<<<(HID * HID + 255) / 256, 256, 0, stream>>>(lp_w, WLT, HID, HID);

  const int NB = HID / (16 * NBLK);          // 8 waves cover all 512 cols of a 16-row stripe
  const int wtiles = (MPAD / 16) * NB;       // 625 * 8 = 5000 waves
  const int gemmBlocks = (wtiles + 7) / 8;   // 8 waves / block

  // GCN layer 1: GEMM + fused gather(agg)+bias+ReLU+bf16
  k_gemm_wmma<NODE_D><<<gemmBlocks, 256, 0, stream>>>(Xbf, W1T, HL, nullptr, NB, wtiles, 0);
  k_gather   <<<MPAD, HID, 0, stream>>>(HL, off, srcid, wnorm, dnv, g1_b, HB);

  // GCN layer 2
  k_gemm_wmma<HID><<<gemmBlocks, 256, 0, stream>>>(HB, W2T, HL, nullptr, NB, wtiles, 0);
  k_gather   <<<MPAD, HID, 0, stream>>>(HL, off, srcid, wnorm, dnv, g2_b, HB);

  // projection: h1 = relu(h @ lp_w + lp_b)  (fused epilogue) -> HA (H1, fp32)
  k_gemm_wmma<HID><<<gemmBlocks, 256, 0, stream>>>(HB, WLT, HA, lp_b, NB, wtiles, 1);

  // backbone means + head
  k_backbone<<<MAX_LIG, HID, 0, stream>>>(HA, pred_pos, bbf);
  k_tail<<<1, HID, 0, stream>>>(HA, bbf, metal_id, pred_pos, emb, mp_w, mp_b,
                                gw1, gb1, gw2, gb2,
                                tq_w, tq_b, tk_w, tk_b, tv_w, tv_b,
                                tsk_w, tsk_b, pr_w1, pr_b1, pr_w2, pr_b2,
                                (float*)d_out);
}
